// LayoutAttention_1305670058070
// MI455X (gfx1250) — matrix operand order repaired and checked by reading, not compiled, not run on hardware
//
#include <hip/hip_runtime.h>
#include <hip/hip_bf16.h>
#include <math.h>

// ---------------------------------------------------------------------------
// MI455X (gfx1250) masked multi-head attention, flash-attention style.
// Heavy math on v_wmma_f32_16x16x32_bf16; K-tile staging via the gfx1250
// async memory->LDS path (ASYNCcnt) with LDS double buffering.
// ---------------------------------------------------------------------------

typedef __attribute__((ext_vector_type(16))) __bf16 v16bf;
typedef __attribute__((ext_vector_type(8)))  float  v8f;

#define NSEQ   4096
#define DMODEL 320
#define HEADS  8
#define DH     40
#define DHP    48          // per-head dim padded to 3x16
#define QKVW   (3*HEADS*DHP)   // 1152
#define AOW    (HEADS*DHP)     // 384
#define SCALE  0.15811388300841897f  // 40^-0.5

// ------------------------------- prep kernels ------------------------------

__global__ void cast_x_kernel(const float* __restrict__ x, __bf16* __restrict__ xb, int n) {
    int i = blockIdx.x * blockDim.x + threadIdx.x;
    if (i < n) xb[i] = (__bf16)x[i];
}

__global__ void codes_kernel(const float* __restrict__ gm, unsigned* __restrict__ codes) {
    int i = blockIdx.x * blockDim.x + threadIdx.x;
    if (i < NSEQ) {
        unsigned c = 0;
        #pragma unroll
        for (int p = 0; p < 3; ++p)
            if (gm[p * NSEQ + i] != 0.0f) c |= (1u << p);
        codes[i] = c;
    }
}

// Wqkv_t[n][k]: n = t*384 + h*48 + d (q|k|v concatenated, dh padded to 48),
// transposed to N-major so B fragments are contiguous in K. Q pre-scaled.
__global__ void build_wqkv_kernel(const float* __restrict__ Wq, const float* __restrict__ Wk,
                                  const float* __restrict__ Wv, __bf16* __restrict__ Wt) {
    int idx = blockIdx.x * blockDim.x + threadIdx.x;
    if (idx >= QKVW * DMODEL) return;
    int n = idx / DMODEL, k = idx % DMODEL;
    int t = n / (HEADS * DHP);
    int r = n % (HEADS * DHP);
    int h = r / DHP, d = r % DHP;
    float v = 0.0f;
    if (d < DH) {
        const float* W = (t == 0) ? Wq : (t == 1) ? Wk : Wv;
        v = W[k * DMODEL + h * DH + d];
        if (t == 0) v *= SCALE;
    }
    Wt[(size_t)n * DMODEL + k] = (__bf16)v;
}

// Wo_t[n][k]: n over 320 output cols, k over padded 384 head-concat dim.
__global__ void build_wo_kernel(const float* __restrict__ Wo, __bf16* __restrict__ Wt) {
    int idx = blockIdx.x * blockDim.x + threadIdx.x;
    if (idx >= DMODEL * AOW) return;
    int n = idx / AOW, k = idx % AOW;
    int h = k / DHP, d = k % DHP;
    float v = (d < DH) ? Wo[(h * DH + d) * DMODEL + n] : 0.0f;
    Wt[(size_t)n * AOW + k] = (__bf16)v;
}

// ------------------------- generic bf16 WMMA GEMM --------------------------
template <bool OUT_F32>
__global__ __launch_bounds__(256) void gemm_bf16_kernel(
    const __bf16* __restrict__ A, const __bf16* __restrict__ Bt,
    int M, int N, int K,
    __bf16* __restrict__ Cb, float* __restrict__ Cf, const float* __restrict__ bias) {
    int lane = threadIdx.x & 31, wid = threadIdx.x >> 5;
    int hi = lane >> 4, lo = lane & 15;
    int m0 = blockIdx.x * 64 + (wid & 3) * 16;
    int n0 = blockIdx.y * 64 + (wid >> 2) * 32;

    v8f acc0 = {}, acc1 = {};
    #pragma unroll 2
    for (int k0 = 0; k0 < K; k0 += 32) {
        union { uint4 u[2]; v16bf v; } af;
        const __bf16* ap = A + (size_t)(m0 + lo) * K + k0 + 16 * hi;
        af.u[0] = *(const uint4*)(ap);
        af.u[1] = *(const uint4*)(ap + 8);
        union { uint4 u[2]; v16bf v; } bf0, bf1;
        const __bf16* bp0 = Bt + (size_t)(n0 + lo) * K + k0 + 16 * hi;
        bf0.u[0] = *(const uint4*)(bp0);
        bf0.u[1] = *(const uint4*)(bp0 + 8);
        const __bf16* bp1 = bp0 + (size_t)16 * K;
        bf1.u[0] = *(const uint4*)(bp1);
        bf1.u[1] = *(const uint4*)(bp1 + 8);
        acc0 = __builtin_amdgcn_wmma_f32_16x16x32_bf16(false, af.v, false, bf0.v, (short)0, acc0, false, false);
        acc1 = __builtin_amdgcn_wmma_f32_16x16x32_bf16(false, af.v, false, bf1.v, (short)0, acc1, false, false);
    }
    #pragma unroll
    for (int r = 0; r < 8; ++r) {
        int m = m0 + r + 8 * hi;
        if (OUT_F32) {
            Cf[(size_t)m * N + n0 + lo]      = acc0[r] + bias[n0 + lo];
            Cf[(size_t)m * N + n0 + 16 + lo] = acc1[r] + bias[n0 + 16 + lo];
        } else {
            Cb[(size_t)m * N + n0 + lo]      = (__bf16)acc0[r];
            Cb[(size_t)m * N + n0 + 16 + lo] = (__bf16)acc1[r];
        }
    }
}

// ---------------------- async global -> LDS staging ------------------------
// gfx1250 GLOBAL_LOAD_ASYNC_TO_LDS_B128 (ASYNCcnt-tracked, no VGPR round
// trip). Parameter types per the compiler diagnostic: pointers to
// int __vector_size(16), global AS source, LDS AS destination.
__device__ __forceinline__ void stage_b128_async(const __bf16* gp, __bf16* lp) {
#if __has_builtin(__builtin_amdgcn_global_load_async_to_lds_b128)
    typedef int v4i_vs __attribute__((vector_size(4 * sizeof(int))));
    typedef __attribute__((address_space(1))) v4i_vs* gas_t;
    typedef __attribute__((address_space(3))) v4i_vs* las_t;
    __builtin_amdgcn_global_load_async_to_lds_b128((gas_t)gp, (las_t)lp, 0, 0);
#else
    *(uint4*)lp = *(const uint4*)gp;
#endif
}
__device__ __forceinline__ void wait_async0() {
    asm volatile("s_wait_asynccnt 0x0" ::: "memory");
}

// --------------------------- flash attention core --------------------------
// Grid: (N/128, HEADS). 8 waves x 16 query rows; K tile double-buffered in
// LDS and filled by async loads; V tile transposed through registers.
#define LDK 72   // K tile pitch (bf16), 144B rows keep 16B alignment
#define LDV 48   // V^T tile pitch
#define LDP 32   // per-wave P scratch pitch

__global__ __launch_bounds__(256) void attn_kernel(
    const __bf16* __restrict__ QKV,        // [N][1152]
    const unsigned* __restrict__ codes,    // [N] 3-bit phase codes
    __bf16* __restrict__ AO) {             // [N][384]
    __shared__ __bf16 Kt[2][32 * LDK];     // double-buffered K tile [key][dh64]
    __shared__ __bf16 Vt[64 * LDV];        // V tile transposed [dh64][key]
    __shared__ __bf16 Ps[8][16 * LDP];     // per-wave P relayout scratch

    int tid = threadIdx.x;
    int lane = tid & 31, wid = tid >> 5;
    int hi = lane >> 4, lo = lane & 15;
    int head = blockIdx.y;
    int q0 = blockIdx.x * 128 + wid * 16;
    int qcol = head * DHP;
    int kcol = HEADS * DHP + head * DHP;
    int vcol = 2 * HEADS * DHP + head * DHP;

    // staging geometry (per thread)
    const int kkey = tid >> 3, kd0 = (tid & 7) * 8;   // K tile: 32 x 8 chunks
    const int vkey = tid & 31, vd0 = (tid >> 5) * 8;  // V tile: 8 chunks x 32
    const bool kvalid = (kd0 < DHP), vvalid = (vd0 < DHP);

    // one-time zero fill of K-tile pad columns (async lanes only write d<48)
    if (tid < 64) {
        int buf = tid >> 5, key = tid & 31;
        *(uint4*)(&Kt[buf][key * LDK + 48]) = uint4{0, 0, 0, 0};
        *(uint4*)(&Kt[buf][key * LDK + 56]) = uint4{0, 0, 0, 0};
    }

    // Q fragments (A operand), K-dim 48 padded to 64 -> two x32 chunks.
    v16bf qa[2];
    #pragma unroll
    for (int c = 0; c < 2; ++c)
        #pragma unroll
        for (int half = 0; half < 2; ++half)
            #pragma unroll
            for (int j = 0; j < 8; ++j) {
                int d = c * 32 + hi * 16 + half * 8 + j;
                __bf16 v = (__bf16)0.0f;
                if (d < DHP) v = QKV[(size_t)(q0 + lo) * QKVW + qcol + d];
                qa[c][half * 8 + j] = v;
            }

    // Pack this lane's 8 query-row codes (rows r + 8*hi) into nibbles.
    unsigned qc = 0;
    #pragma unroll
    for (int r = 0; r < 8; ++r) qc |= (codes[q0 + 8 * hi + r] & 7u) << (4 * r);

    float mrow[8], lrow[8];
    v8f o0 = {}, o1 = {}, o2 = {};
    #pragma unroll
    for (int r = 0; r < 8; ++r) { mrow[r] = -INFINITY; lrow[r] = 0.0f; }

    // prologue: async K tile 0 into buffer 0; V tile 0 into registers
    if (kvalid)
        stage_b128_async(QKV + (size_t)kkey * QKVW + kcol + kd0, &Kt[0][kkey * LDK + kd0]);
    uint4 vreg = {0, 0, 0, 0};
    if (vvalid) vreg = *(const uint4*)(QKV + (size_t)vkey * QKVW + vcol + vd0);

    for (int kb = 0; kb < NSEQ; kb += 32) {
        const int p = (kb >> 5) & 1;
        // commit V registers to LDS (transpose scatter)
        {
            const __bf16* pv = (const __bf16*)&vreg;
            #pragma unroll
            for (int j = 0; j < 8; ++j)
                Vt[(vd0 + j) * LDV + vkey] = vvalid ? pv[j] : (__bf16)0.0f;
        }
        wait_async0();          // this wave's K-tile async writes landed
        __syncthreads();        // everyone's K async + V stores visible

        // issue async K loads for the NEXT tile into the other buffer,
        // and prefetch the next V tile into registers, while computing.
        if (kb + 32 < NSEQ) {
            if (kvalid)
                stage_b128_async(QKV + (size_t)(kb + 32 + kkey) * QKVW + kcol + kd0,
                                 &Kt[1 - p][kkey * LDK + kd0]);
            if (vvalid) vreg = *(const uint4*)(QKV + (size_t)(kb + 32 + vkey) * QKVW + vcol + vd0);
        }

        // S(16q x 32k) = Q @ K^T : preload all 4 B fragments, then 4 WMMA
        const __bf16* Kb = Kt[p];
        union { uint4 u[2]; v16bf v; } b00, b10, b01, b11;
        {
            const __bf16* r0 = &Kb[lo * LDK + 16 * hi];
            const __bf16* r1 = &Kb[(16 + lo) * LDK + 16 * hi];
            b00.u[0] = *(const uint4*)(r0);      b00.u[1] = *(const uint4*)(r0 + 8);
            b10.u[0] = *(const uint4*)(r1);      b10.u[1] = *(const uint4*)(r1 + 8);
            b01.u[0] = *(const uint4*)(r0 + 32); b01.u[1] = *(const uint4*)(r0 + 40);
            b11.u[0] = *(const uint4*)(r1 + 32); b11.u[1] = *(const uint4*)(r1 + 40);
        }
        v8f s0 = {}, s1 = {};
        s0 = __builtin_amdgcn_wmma_f32_16x16x32_bf16(false, qa[0], false, b00.v, (short)0, s0, false, false);
        s1 = __builtin_amdgcn_wmma_f32_16x16x32_bf16(false, qa[0], false, b10.v, (short)0, s1, false, false);
        s0 = __builtin_amdgcn_wmma_f32_16x16x32_bf16(false, qa[1], false, b01.v, (short)0, s0, false, false);
        s1 = __builtin_amdgcn_wmma_f32_16x16x32_bf16(false, qa[1], false, b11.v, (short)0, s1, false, false);

        // mask + online softmax (C layout: col = lane&15, row = r + 8*hi)
        unsigned kc0 = codes[kb + lo], kc1 = codes[kb + 16 + lo];
        #pragma unroll
        for (int r = 0; r < 8; ++r) {
            unsigned qn = (qc >> (4 * r)) & 15u;
            float a0 = ((qn & kc0) == 0u) ? -1e30f : s0[r];
            float a1 = ((qn & kc1) == 0u) ? -1e30f : s1[r];
            float t = fmaxf(a0, a1);
            t = fmaxf(t, __shfl_xor(t, 1, 32));
            t = fmaxf(t, __shfl_xor(t, 2, 32));
            t = fmaxf(t, __shfl_xor(t, 4, 32));
            t = fmaxf(t, __shfl_xor(t, 8, 32));
            float mn = fmaxf(mrow[r], t);
            float alpha = __expf(mrow[r] - mn);
            mrow[r] = mn;
            float p0 = __expf(a0 - mn);
            float p1 = __expf(a1 - mn);
            s0[r] = p0;
            s1[r] = p1;
            float ls = p0 + p1;
            ls += __shfl_xor(ls, 1, 32);
            ls += __shfl_xor(ls, 2, 32);
            ls += __shfl_xor(ls, 4, 32);
            ls += __shfl_xor(ls, 8, 32);
            lrow[r] = lrow[r] * alpha + ls;
            o0[r] *= alpha; o1[r] *= alpha; o2[r] *= alpha;
        }

        // P: C layout -> A layout via per-wave LDS scratch (wave-local DS
        // ordering only: DS ops of one wave complete in order).
        __bf16* pw = Ps[wid];
        #pragma unroll
        for (int r = 0; r < 8; ++r) {
            pw[(r + 8 * hi) * LDP + lo]      = (__bf16)s0[r];
            pw[(r + 8 * hi) * LDP + 16 + lo] = (__bf16)s1[r];
        }
        asm volatile("s_wait_dscnt 0x0" ::: "memory");
        union { uint4 u[2]; v16bf v; } pa;
        const __bf16* pp = pw + lo * LDP + 16 * hi;
        pa.u[0] = *(const uint4*)(pp);
        pa.u[1] = *(const uint4*)(pp + 8);

        // O(16q x 48dh) += P @ V : preload 3 B fragments, then 3 WMMA
        union { uint4 u[2]; v16bf v; } bv0, bv1, bv2;
        const __bf16* vp0 = &Vt[(0 * 16 + lo) * LDV + 16 * hi];
        const __bf16* vp1 = &Vt[(1 * 16 + lo) * LDV + 16 * hi];
        const __bf16* vp2 = &Vt[(2 * 16 + lo) * LDV + 16 * hi];
        bv0.u[0] = *(const uint4*)(vp0); bv0.u[1] = *(const uint4*)(vp0 + 8);
        bv1.u[0] = *(const uint4*)(vp1); bv1.u[1] = *(const uint4*)(vp1 + 8);
        bv2.u[0] = *(const uint4*)(vp2); bv2.u[1] = *(const uint4*)(vp2 + 8);
        o0 = __builtin_amdgcn_wmma_f32_16x16x32_bf16(false, pa.v, false, bv0.v, (short)0, o0, false, false);
        o1 = __builtin_amdgcn_wmma_f32_16x16x32_bf16(false, pa.v, false, bv1.v, (short)0, o1, false, false);
        o2 = __builtin_amdgcn_wmma_f32_16x16x32_bf16(false, pa.v, false, bv2.v, (short)0, o2, false, false);

        __syncthreads();   // all K/V fragment reads done before restaging
    }

    // normalize and emit bf16 head outputs (padded cols are exact zeros)
    #pragma unroll
    for (int r = 0; r < 8; ++r) {
        float inv = 1.0f / lrow[r];
        int m = q0 + r + 8 * hi;
        __bf16* ao = AO + (size_t)m * AOW + head * DHP;
        ao[lo]      = (__bf16)(o0[r] * inv);
        ao[16 + lo] = (__bf16)(o1[r] * inv);
        ao[32 + lo] = (__bf16)(o2[r] * inv);
    }
}

// ------------------------------- launcher ----------------------------------

static size_t align_up(size_t v, size_t a) { return (v + a - 1) & ~(a - 1); }

extern "C" void kernel_launch(void* const* d_in, const int* in_sizes, int n_in,
                              void* d_out, int out_size, void* d_ws, size_t ws_size,
                              hipStream_t stream) {
    const float* x   = (const float*)d_in[0];
    const float* gm  = (const float*)d_in[1];
    const float* Wq  = (const float*)d_in[2];
    const float* Wk  = (const float*)d_in[3];
    const float* Wv  = (const float*)d_in[4];
    const float* Wo  = (const float*)d_in[5];
    const float* bo  = (const float*)d_in[6];
    float* out = (float*)d_out;

    // workspace carve-up (~16.2 MB total)
    char* ws = (char*)d_ws;
    size_t off = 0;
    __bf16* Xb      = (__bf16*)(ws + off); off = align_up(off + (size_t)NSEQ * DMODEL * 2, 256);
    __bf16* Wqkv_t  = (__bf16*)(ws + off); off = align_up(off + (size_t)QKVW * DMODEL * 2, 256);
    __bf16* QKV     = (__bf16*)(ws + off); off = align_up(off + (size_t)NSEQ * QKVW * 2, 256);
    unsigned* codes = (unsigned*)(ws + off); off = align_up(off + (size_t)NSEQ * 4, 256);
    __bf16* AO      = (__bf16*)(ws + off); off = align_up(off + (size_t)NSEQ * AOW * 2, 256);
    __bf16* Wo_t    = (__bf16*)(ws + off); off = align_up(off + (size_t)DMODEL * AOW * 2, 256);
    (void)ws_size; (void)in_sizes; (void)n_in; (void)out_size;

    // prep
    cast_x_kernel<<<(NSEQ * DMODEL + 255) / 256, 256, 0, stream>>>(x, Xb, NSEQ * DMODEL);
    codes_kernel<<<(NSEQ + 255) / 256, 256, 0, stream>>>(gm, codes);
    build_wqkv_kernel<<<((size_t)QKVW * DMODEL + 255) / 256, 256, 0, stream>>>(Wq, Wk, Wv, Wqkv_t);
    build_wo_kernel<<<((size_t)DMODEL * AOW + 255) / 256, 256, 0, stream>>>(Wo, Wo_t);

    // QKV projection: (4096x320) @ (320x1152) -> bf16 QKV
    gemm_bf16_kernel<false><<<dim3(NSEQ / 64, QKVW / 64), 256, 0, stream>>>(
        Xb, Wqkv_t, NSEQ, QKVW, DMODEL, QKV, nullptr, nullptr);

    // flash attention over all heads
    attn_kernel<<<dim3(NSEQ / 128, HEADS), 256, 0, stream>>>(QKV, codes, AO);

    // output projection: (4096x384) @ (384x320) + bias -> f32 out
    gemm_bf16_kernel<true><<<dim3(NSEQ / 64, DMODEL / 64), 256, 0, stream>>>(
        AO, Wo_t, NSEQ, DMODEL, AOW, nullptr, out, bo);
}
